// SwinBlock_87943750353537
// MI455X (gfx1250) — matrix operand order repaired and checked by reading, not compiled, not run on hardware
//
#include <hip/hip_runtime.h>
#include <math.h>

// ---------------------------------------------------------------------------
// Swin block forward for MI455X (gfx1250, wave32, WMMA bf16 16x16x32 + TDM).
// B=32 H=W=56 C=384 nH=12 hd=32 ws=7 N=49(pad 64) nW=64, M = 32*64*49 = 100352
// ---------------------------------------------------------------------------

typedef __attribute__((ext_vector_type(16))) __bf16 v16bf;
typedef __attribute__((ext_vector_type(8)))  float  v8f;
typedef unsigned int u32x4 __attribute__((ext_vector_type(4)));
typedef int          i32x8 __attribute__((ext_vector_type(8)));
typedef int          i32x4 __attribute__((ext_vector_type(4)));

#define M_ROWS 100352   // B * nW * N
#define C_DIM  384
#define HID    1536
#define QKV_N  1152

#if defined(__gfx1250__) && __has_builtin(__builtin_amdgcn_tensor_load_to_lds)
#define HAVE_TDM 1
#else
#define HAVE_TDM 0
#endif

__device__ inline v8f wmma_bf16(v16bf a, v16bf b, v8f c) {
  return __builtin_amdgcn_wmma_f32_16x16x32_bf16(false, a, false, b, (short)0, c,
                                                 false, false);
}

// A-matrix fragment (16x32 bf16), ISA 7.12.2: lane m=lane&15;
// lanes<16 hold K=0..7,16..23 ; lanes>=16 hold K=8..15,24..31 (2 halves/VGPR).
__device__ inline v16bf load_frag_a(const __bf16* p, int ld, int lane) {
  int m  = lane & 15;
  int kb = (lane < 16) ? 0 : 8;
  const uint4* lo = reinterpret_cast<const uint4*>(p + m * ld + kb);
  const uint4* hi = reinterpret_cast<const uint4*>(p + m * ld + 16 + kb);
  union { v16bf v; uint4 u[2]; } u;
  u.u[0] = lo[0];
  u.u[1] = hi[0];
  return u.v;
}

// B-matrix fragment (32x16 bf16, K x N) from LDS stored N-major (row n = 32 K's).
__device__ inline v16bf load_frag_b(const __bf16* p, int ld, int lane) {
  int n  = lane & 15;
  int kb = (lane < 16) ? 0 : 8;
  const uint4* q = reinterpret_cast<const uint4*>(p + n * ld + kb);
  union { v16bf v; uint4 u[2]; } u;
  u.u[0] = q[0];
  u.u[1] = q[2];
  return u.v;
}

#if HAVE_TDM
// Raw LDS byte offset = low 32 bits of the flat shared address (ISA 10.2).
__device__ inline unsigned lds_byte_off(const void* p) {
  return (unsigned)(unsigned long long)(uintptr_t)p;
}

// Issue a TDM 2D tile load: bf16 elements, tile (tile_d0 x tile_d1),
// logical extent (tens_d0 x tens_d1) relative to tile start (OOB -> 0),
// row stride `stride0` elements. Groups 2/3 zero (2D tensor).
__device__ inline void tdm_load_2d(unsigned lds_off, const void* gaddr,
                                   unsigned tile_d0, unsigned tile_d1,
                                   unsigned tens_d0, unsigned tens_d1,
                                   unsigned stride0) {
  unsigned long long ga = (unsigned long long)(uintptr_t)gaddr;
  u32x4 g0;
  g0[0] = 1u;                                                // count=1, user D#
  g0[1] = lds_off;                                           // LDS byte address
  g0[2] = (unsigned)ga;                                      // global addr [31:0]
  g0[3] = (unsigned)((ga >> 32) & 0x1FFFFFFu) | (2u << 30);  // addr[56:32]|type=2
  i32x8 g1;
  g1[0] = (int)(1u << 16);                                   // data_size=2B
  g1[1] = (int)((tens_d0 & 0xFFFFu) << 16);                  // tensor_dim0 lo
  g1[2] = (int)((tens_d0 >> 16) | ((tens_d1 & 0xFFFFu) << 16));
  g1[3] = (int)(((tens_d1 >> 16) & 0xFFFFu) | (tile_d0 << 16));
  g1[4] = (int)(tile_d1 & 0xFFFFu);                          // tile_dim1, dim2=0
  g1[5] = (int)stride0;                                      // dim0 stride lo
  g1[6] = 0;
  g1[7] = 0;
  i32x4 gz = {0, 0, 0, 0};
#if __clang_major__ >= 23
  i32x8 gz8 = {0, 0, 0, 0, 0, 0, 0, 0};
  __builtin_amdgcn_tensor_load_to_lds(g0, g1, gz, gz, gz8, 0);
#else
  __builtin_amdgcn_tensor_load_to_lds(g0, g1, gz, gz, 0);
#endif
}
#endif  // HAVE_TDM

// ---------------------------------------------------------------------------
// Weight convert + transpose: src f32 (K x N) -> dst bf16 (N x K)
// ---------------------------------------------------------------------------
__global__ __launch_bounds__(256) void wcvt_kernel(const float* __restrict__ src,
                                                   __bf16* __restrict__ dst,
                                                   int K, int N) {
  int tid = blockIdx.x * 256 + threadIdx.x;
  if (tid >= K * N) return;
  int k = tid / N, n = tid - k * N;
  dst[(size_t)n * K + k] = (__bf16)src[tid];
}

// ---------------------------------------------------------------------------
// Fused relative-position-bias + shifted-window mask table.
// T[((w*12+head)<<12) + row*64 + col], padded rows -> 0, padded cols -> -1e30.
// ---------------------------------------------------------------------------
__global__ __launch_bounds__(256) void biasmask_kernel(const float* __restrict__ tbl,
                                                       float* __restrict__ T) {
  int tid = blockIdx.x * 256 + threadIdx.x;   // < 64*12*4096
  int col  = tid & 63;
  int row  = (tid >> 6) & 63;
  int hw   = tid >> 12;
  int head = hw % 12;
  int w    = hw / 12;
  float v;
  if (row >= 49) {
    v = 0.0f;
  } else if (col >= 49) {
    v = -1.0e30f;
  } else {
    int i1 = row / 7, j1 = row - i1 * 7;
    int i2 = col / 7, j2 = col - i2 * 7;
    int rel = (i1 - i2 + 6) * 13 + (j1 - j2 + 6);
    float bias = tbl[rel * 12 + head];
    int wh = w >> 3, ww = w & 7;
    auto reg = [](int a) { return (a < 49) ? 0 : ((a < 53) ? 1 : 2); };
    int r1 = reg(wh * 7 + i1) * 3 + reg(ww * 7 + j1);
    int r2 = reg(wh * 7 + i2) * 3 + reg(ww * 7 + j2);
    v = bias + ((r1 != r2) ? -100.0f : 0.0f);
  }
  T[tid] = v;
}

// ---------------------------------------------------------------------------
// LayerNorm kernels: one wave per token row, 12 channels/lane.
// ---------------------------------------------------------------------------
__device__ inline void ln_row(const float* __restrict__ xr,
                              const float* __restrict__ g,
                              const float* __restrict__ bb,
                              __bf16* __restrict__ orow, int lane) {
  float v[12];
  float s = 0.0f;
#pragma unroll
  for (int t = 0; t < 12; ++t) { v[t] = xr[t * 32 + lane]; s += v[t]; }
#pragma unroll
  for (int off = 16; off > 0; off >>= 1) s += __shfl_xor(s, off, 32);
  float mean = s * (1.0f / 384.0f);
  float q = 0.0f;
#pragma unroll
  for (int t = 0; t < 12; ++t) { float d = v[t] - mean; q += d * d; }
#pragma unroll
  for (int off = 16; off > 0; off >>= 1) q += __shfl_xor(q, off, 32);
  float inv = rsqrtf(q * (1.0f / 384.0f) + 1e-5f);
#pragma unroll
  for (int t = 0; t < 12; ++t) {
    int c = t * 32 + lane;
    orow[c] = (__bf16)((v[t] - mean) * inv * g[c] + bb[c]);
  }
}

__global__ __launch_bounds__(256) void ln1_win_kernel(const float* __restrict__ x,
                                                      const float* __restrict__ g,
                                                      const float* __restrict__ bb,
                                                      __bf16* __restrict__ out) {
  int r    = blockIdx.x * 8 + (threadIdx.x >> 5);
  int lane = threadIdx.x & 31;
  int b  = r / 3136;
  int rr = r - b * 3136;
  int w  = rr / 49;
  int n  = rr - w * 49;
  int i = n / 7, j = n - i * 7;
  int sh = (w >> 3) * 7 + i + 3; if (sh >= 56) sh -= 56;
  int sw = (w & 7)  * 7 + j + 3; if (sw >= 56) sw -= 56;
  const float* xr = x + ((size_t)b * 3136 + sh * 56 + sw) * C_DIM;
  ln_row(xr, g, bb, out + (size_t)r * C_DIM, lane);
}

__global__ __launch_bounds__(256) void ln2_kernel(const float* __restrict__ x,
                                                  const float* __restrict__ g,
                                                  const float* __restrict__ bb,
                                                  __bf16* __restrict__ out) {
  int r    = blockIdx.x * 8 + (threadIdx.x >> 5);
  int lane = threadIdx.x & 31;
  ln_row(x + (size_t)r * C_DIM, g, bb, out + (size_t)r * C_DIM, lane);
}

// ---------------------------------------------------------------------------
// Generic WMMA bf16 GEMM: out(MxN) = A(MxK) * B(KxN) + bias, templated epilogue.
// Block: 256 threads = 8 waves (4x2), block tile 128x64, K step 32.
// TDM path: double-buffered LDS, wave 0 DMAs next tiles while all waves WMMA.
// EPI 0: bf16 bias | 1: bf16 bias+GELU | 2: f32 bias+resid | 3: 2 + win-reverse
// ---------------------------------------------------------------------------
template <int EPI>
__global__ __launch_bounds__(256) void gemm_bf16(const __bf16* __restrict__ A,
                                                 const __bf16* __restrict__ Bt,
                                                 const float* __restrict__ bias,
                                                 int K, int N,
                                                 __bf16* __restrict__ out_bf,
                                                 float* __restrict__ out_f,
                                                 const float* __restrict__ resid) {
  const int tid  = threadIdx.x;
  const int lane = tid & 31;
  const int wv   = tid >> 5;
  const int mw   = wv & 3;            // 0..3 -> 32-row slab
  const int nw   = wv >> 2;           // 0..1 -> 32-col slab
  const int m0   = blockIdx.y * 128;
  const int n0   = blockIdx.x * 64;

  v8f acc[2][2] = {};

#if HAVE_TDM
  __shared__ __align__(16) __bf16 As[2][128 * 32];
  __shared__ __align__(16) __bf16 Bs[2][64 * 32];
  const bool issuer = (wv == 0);
  if (issuer) {  // prime buffer 0
    tdm_load_2d(lds_byte_off(&As[0][0]), A + (size_t)m0 * K, 32, 128, 32, 128, K);
    tdm_load_2d(lds_byte_off(&Bs[0][0]), Bt + (size_t)n0 * K, 32, 64, 32, 64, K);
  }
  for (int k0 = 0; k0 < K; k0 += 32) {
    const int cur = (k0 >> 5) & 1;
    if (issuer) {
      if (k0 + 32 < K) {  // DMA next tiles, then wait for current pair (in-order)
        const int nxt = cur ^ 1;
        tdm_load_2d(lds_byte_off(&As[nxt][0]),
                    A + (size_t)m0 * K + k0 + 32, 32, 128, 32, 128, K);
        tdm_load_2d(lds_byte_off(&Bs[nxt][0]),
                    Bt + (size_t)n0 * K + k0 + 32, 32, 64, 32, 64, K);
        __builtin_amdgcn_s_wait_tensorcnt(2);
      } else {
        __builtin_amdgcn_s_wait_tensorcnt(0);
      }
    }
    __syncthreads();

    const __bf16* Ab = &As[cur][0];
    const __bf16* Bb = &Bs[cur][0];
    v16bf a0 = load_frag_a(Ab + (mw * 32 + 0)  * 32, 32, lane);
    v16bf a1 = load_frag_a(Ab + (mw * 32 + 16) * 32, 32, lane);
    v16bf b0 = load_frag_b(Bb + (nw * 32 + 0)  * 32, 32, lane);
    v16bf b1 = load_frag_b(Bb + (nw * 32 + 16) * 32, 32, lane);
    acc[0][0] = wmma_bf16(a0, b0, acc[0][0]);
    acc[0][1] = wmma_bf16(a0, b1, acc[0][1]);
    acc[1][0] = wmma_bf16(a1, b0, acc[1][0]);
    acc[1][1] = wmma_bf16(a1, b1, acc[1][1]);
    __syncthreads();  // reads of cur done before it becomes the DMA target
  }
#else
  __shared__ __align__(16) __bf16 As[128 * 32];
  __shared__ __align__(16) __bf16 Bs[64 * 32];
  const int arow = tid >> 1, ahalf = tid & 1;
  const int brow = tid >> 2, bq    = tid & 3;
  for (int k0 = 0; k0 < K; k0 += 32) {
    __syncthreads();
    {
      const uint4* sa = reinterpret_cast<const uint4*>(
          A + (size_t)(m0 + arow) * K + k0 + ahalf * 16);
      uint4* da = reinterpret_cast<uint4*>(As + arow * 32 + ahalf * 16);
      da[0] = sa[0];
      da[1] = sa[1];
      const uint4* sb = reinterpret_cast<const uint4*>(
          Bt + (size_t)(n0 + brow) * K + k0 + bq * 8);
      reinterpret_cast<uint4*>(Bs + brow * 32 + bq * 8)[0] = sb[0];
    }
    if (k0 + 32 < K) {
      __builtin_prefetch((const void*)(A + (size_t)(m0 + arow) * K + k0 + 32 + ahalf * 16), 0, 1);
      __builtin_prefetch((const void*)(Bt + (size_t)(n0 + brow) * K + k0 + 32 + bq * 8), 0, 1);
    }
    __syncthreads();
    v16bf a0 = load_frag_a(As + (mw * 32 + 0)  * 32, 32, lane);
    v16bf a1 = load_frag_a(As + (mw * 32 + 16) * 32, 32, lane);
    v16bf b0 = load_frag_b(Bs + (nw * 32 + 0)  * 32, 32, lane);
    v16bf b1 = load_frag_b(Bs + (nw * 32 + 16) * 32, 32, lane);
    acc[0][0] = wmma_bf16(a0, b0, acc[0][0]);
    acc[0][1] = wmma_bf16(a0, b1, acc[0][1]);
    acc[1][0] = wmma_bf16(a1, b0, acc[1][0]);
    acc[1][1] = wmma_bf16(a1, b1, acc[1][1]);
  }
#endif

#pragma unroll
  for (int sm = 0; sm < 2; ++sm) {
#pragma unroll
    for (int sn = 0; sn < 2; ++sn) {
      int col = n0 + nw * 32 + sn * 16 + (lane & 15);
      float bv = bias[col];
#pragma unroll
      for (int r = 0; r < 8; ++r) {
        int row = m0 + mw * 32 + sm * 16 + r + ((lane < 16) ? 0 : 8);
        float v = acc[sm][sn][r] + bv;
        if (EPI == 0) {
          out_bf[(size_t)row * N + col] = (__bf16)v;
        } else if (EPI == 1) {  // exact GELU
          float gl = v * 0.5f * (1.0f + erff(v * 0.70710678118654752f));
          out_bf[(size_t)row * N + col] = (__bf16)gl;
        } else if (EPI == 2) {
          size_t o = (size_t)row * N + col;
          out_f[o] = resid[o] + v;
        } else {  // EPI 3: window reverse + un-shift + residual (N==384)
          int b  = row / 3136;
          int rr = row - b * 3136;
          int w  = rr / 49;
          int n  = rr - w * 49;
          int i = n / 7, j = n - i * 7;
          int hh = (w >> 3) * 7 + i + 3; if (hh >= 56) hh -= 56;
          int ww = (w & 7)  * 7 + j + 3; if (ww >= 56) ww -= 56;
          size_t o = ((size_t)b * 3136 + hh * 56 + ww) * 384 + col;
          out_f[o] = resid[o] + v;
        }
      }
    }
  }
}

// ---------------------------------------------------------------------------
// Fused windowed attention: one wave per (window, head), 2 waves per block.
// Q/K tiles DMA'd by TDM (OOB rows auto-zero), V^T staged manually (transpose).
// S = scale*Q*K^T + T ; softmax in registers (C/D lane layout) ; O = P*V.
// ---------------------------------------------------------------------------
__global__ __launch_bounds__(64) void attn_kernel(const __bf16* __restrict__ qkv,
                                                  const float* __restrict__ Tt,
                                                  __bf16* __restrict__ attn_out) {
  // per-wave LDS slice: Qs 64x32 | Ks 64x32 | Vt 32x64 | Ps 64x64  (bf16)
  __shared__ __align__(16) __bf16 smem[2 * 10240];
  const int lane = threadIdx.x & 31;
  const int wv   = threadIdx.x >> 5;
  __bf16* Qs = smem + wv * 10240;
  __bf16* Ks = Qs + 2048;
  __bf16* Vt = Qs + 4096;
  __bf16* Ps = Qs + 6144;

  const int gwh  = blockIdx.x * 2 + wv;   // window-head id
  const int head = gwh % 12;
  const int win  = gwh / 12;              // 0 .. 2047
  const size_t row0 = (size_t)win * 49;
  const int qoff = head * 32, koff = 384 + head * 32, voff = 768 + head * 32;

#if HAVE_TDM
  // per-wave DMA of Q and K 64x32 tiles; rows 49..63 are OOB -> zero-filled
  tdm_load_2d(lds_byte_off(Qs), qkv + row0 * QKV_N + qoff, 32, 64, 32, 49, QKV_N);
  tdm_load_2d(lds_byte_off(Ks), qkv + row0 * QKV_N + koff, 32, 64, 32, 49, QKV_N);
#else
  for (int t = lane; t < 64; t += 32) {
    uint4 z = {0u, 0u, 0u, 0u};
    uint4 qv[4] = {z, z, z, z}, kv[4] = {z, z, z, z};
    if (t < 49) {
      const uint4* qp = reinterpret_cast<const uint4*>(qkv + (row0 + t) * QKV_N + qoff);
      const uint4* kp = reinterpret_cast<const uint4*>(qkv + (row0 + t) * QKV_N + koff);
#pragma unroll
      for (int c = 0; c < 4; ++c) { qv[c] = qp[c]; kv[c] = kp[c]; }
    }
    uint4* qd = reinterpret_cast<uint4*>(Qs + t * 32);
    uint4* kd = reinterpret_cast<uint4*>(Ks + t * 32);
#pragma unroll
    for (int c = 0; c < 4; ++c) { qd[c] = qv[c]; kd[c] = kv[c]; }
  }
#endif
  // V^T staging overlaps with the in-flight TDM transfers
  for (int t = lane; t < 64; t += 32) {
    if (t < 49) {
      const __bf16* vp = qkv + (row0 + t) * QKV_N + voff;
#pragma unroll
      for (int c = 0; c < 32; ++c) Vt[c * 64 + t] = vp[c];
    } else {
#pragma unroll
      for (int c = 0; c < 32; ++c) Vt[c * 64 + t] = (__bf16)0.0f;
    }
  }
#if HAVE_TDM
  __builtin_amdgcn_s_wait_tensorcnt(0);
#endif
  __syncthreads();

  // ---- S = Q * K^T : 4x4 tiles, K-dim = hd = 32 = one WMMA ----
  v16bf qa[4];
#pragma unroll
  for (int mt = 0; mt < 4; ++mt) qa[mt] = load_frag_a(Qs + mt * 16 * 32, 32, lane);
  v8f sacc[4][4] = {};
#pragma unroll
  for (int nt = 0; nt < 4; ++nt) {
    v16bf kb = load_frag_b(Ks + nt * 16 * 32, 32, lane);
#pragma unroll
    for (int mt = 0; mt < 4; ++mt) sacc[mt][nt] = wmma_bf16(qa[mt], kb, sacc[mt][nt]);
  }

  // ---- softmax in registers (rows live in 16-lane halves across nt tiles) ----
  const float scale = 0.17677669529663687f;  // 1/sqrt(32)
  const float* Tb = Tt + (((size_t)(win & 63) * 12 + head) << 12);
#pragma unroll
  for (int mt = 0; mt < 4; ++mt) {
#pragma unroll
    for (int r = 0; r < 8; ++r) {
      int row  = mt * 16 + r + ((lane < 16) ? 0 : 8);
      int colb = lane & 15;
      float sv[4];
      float mx = -3.0e38f;
#pragma unroll
      for (int nt = 0; nt < 4; ++nt) {
        sv[nt] = sacc[mt][nt][r] * scale + Tb[row * 64 + nt * 16 + colb];
        mx = fmaxf(mx, sv[nt]);
      }
#pragma unroll
      for (int off = 8; off > 0; off >>= 1) mx = fmaxf(mx, __shfl_xor(mx, off, 32));
      float sum = 0.0f;
#pragma unroll
      for (int nt = 0; nt < 4; ++nt) { sv[nt] = __expf(sv[nt] - mx); sum += sv[nt]; }
#pragma unroll
      for (int off = 8; off > 0; off >>= 1) sum += __shfl_xor(sum, off, 32);
      float inv = 1.0f / sum;
#pragma unroll
      for (int nt = 0; nt < 4; ++nt)
        Ps[row * 64 + nt * 16 + colb] = (__bf16)(sv[nt] * inv);
    }
  }
  __syncthreads();

  // ---- O = P * V : 4x2 tiles, K = 64 in 2 steps ----
  v8f oacc[4][2] = {};
#pragma unroll
  for (int kt = 0; kt < 2; ++kt) {
    v16bf pa[4];
#pragma unroll
    for (int mt = 0; mt < 4; ++mt)
      pa[mt] = load_frag_a(Ps + mt * 16 * 64 + kt * 32, 64, lane);
    v16bf vb[2];
#pragma unroll
    for (int nv = 0; nv < 2; ++nv)
      vb[nv] = load_frag_b(Vt + nv * 16 * 64 + kt * 32, 64, lane);
#pragma unroll
    for (int mt = 0; mt < 4; ++mt)
#pragma unroll
      for (int nv = 0; nv < 2; ++nv)
        oacc[mt][nv] = wmma_bf16(pa[mt], vb[nv], oacc[mt][nv]);
  }

#pragma unroll
  for (int mt = 0; mt < 4; ++mt)
#pragma unroll
    for (int nv = 0; nv < 2; ++nv)
#pragma unroll
      for (int r = 0; r < 8; ++r) {
        int row = mt * 16 + r + ((lane < 16) ? 0 : 8);
        if (row < 49) {
          int col = nv * 16 + (lane & 15);
          attn_out[(row0 + row) * C_DIM + head * 32 + col] = (__bf16)oacc[mt][nv][r];
        }
      }
}

// ---------------------------------------------------------------------------
extern "C" void kernel_launch(void* const* d_in, const int* in_sizes, int n_in,
                              void* d_out, int out_size, void* d_ws, size_t ws_size,
                              hipStream_t stream) {
  (void)in_sizes; (void)n_in; (void)out_size; (void)ws_size;
  const float* x      = (const float*)d_in[0];
  const float* relt   = (const float*)d_in[1];
  const float* qkv_w  = (const float*)d_in[2];
  const float* qkv_b  = (const float*)d_in[3];
  const float* proj_w = (const float*)d_in[4];
  const float* proj_b = (const float*)d_in[5];
  const float* n1w    = (const float*)d_in[6];
  const float* n1b    = (const float*)d_in[7];
  const float* n2w    = (const float*)d_in[8];
  const float* n2b    = (const float*)d_in[9];
  const float* w1     = (const float*)d_in[10];
  const float* b1     = (const float*)d_in[11];
  const float* w2     = (const float*)d_in[12];
  const float* b2     = (const float*)d_in[13];
  float* out = (float*)d_out;

  char* wsb = (char*)d_ws;
  size_t off = 0;
  auto alloc = [&](size_t bytes) {
    size_t o = off;
    off = (off + bytes + 255) & ~(size_t)255;
    return o;
  };
  __bf16* Wqkv  = (__bf16*)(wsb + alloc((size_t)QKV_N * C_DIM * 2));
  __bf16* Wproj = (__bf16*)(wsb + alloc((size_t)C_DIM * C_DIM * 2));
  __bf16* W1t   = (__bf16*)(wsb + alloc((size_t)HID * C_DIM * 2));
  __bf16* W2t   = (__bf16*)(wsb + alloc((size_t)C_DIM * HID * 2));
  float*  T     = (float*)(wsb + alloc((size_t)64 * 12 * 64 * 64 * 4));
  __bf16* LN    = (__bf16*)(wsb + alloc((size_t)M_ROWS * C_DIM * 2));
  __bf16* ATT   = (__bf16*)(wsb + alloc((size_t)M_ROWS * C_DIM * 2));
  __bf16* BIG   = (__bf16*)(wsb + alloc((size_t)M_ROWS * HID * 2));  // qkv then hidden

  // weights -> bf16 transposed (N x K)
  wcvt_kernel<<<(C_DIM * QKV_N + 255) / 256, 256, 0, stream>>>(qkv_w, Wqkv, C_DIM, QKV_N);
  wcvt_kernel<<<(C_DIM * C_DIM + 255) / 256, 256, 0, stream>>>(proj_w, Wproj, C_DIM, C_DIM);
  wcvt_kernel<<<(C_DIM * HID + 255) / 256, 256, 0, stream>>>(w1, W1t, C_DIM, HID);
  wcvt_kernel<<<(HID * C_DIM + 255) / 256, 256, 0, stream>>>(w2, W2t, HID, C_DIM);

  // fused bias+mask table
  biasmask_kernel<<<12288, 256, 0, stream>>>(relt, T);

  // LN1 + shift + window partition
  ln1_win_kernel<<<M_ROWS / 8, 256, 0, stream>>>(x, n1w, n1b, LN);

  // QKV GEMM: (100352 x 384) * (384 x 1152)
  gemm_bf16<0><<<dim3(QKV_N / 64, M_ROWS / 128), 256, 0, stream>>>(
      LN, Wqkv, qkv_b, C_DIM, QKV_N, BIG, nullptr, nullptr);

  // fused windowed attention (24576 window-heads, 2 per block)
  attn_kernel<<<12288, 64, 0, stream>>>(BIG, T, ATT);

  // proj GEMM + window-reverse + un-shift + residual -> d_out (f32)
  gemm_bf16<3><<<dim3(C_DIM / 64, M_ROWS / 128), 256, 0, stream>>>(
      ATT, Wproj, proj_b, C_DIM, C_DIM, nullptr, out, x);

  // LN2 (token order)
  ln2_kernel<<<M_ROWS / 8, 256, 0, stream>>>(out, n2w, n2b, LN);

  // MLP fc1 + GELU
  gemm_bf16<1><<<dim3(HID / 64, M_ROWS / 128), 256, 0, stream>>>(
      LN, W1t, b1, C_DIM, HID, BIG, nullptr, nullptr);

  // MLP fc2 + residual (reads and writes d_out, each element exactly once)
  gemm_bf16<2><<<dim3(C_DIM / 64, M_ROWS / 128), 256, 0, stream>>>(
      BIG, W2t, b2, HID, C_DIM, nullptr, out, out);
}
